// OmniDynamicSeekerAdapter_76991583748288
// MI455X (gfx1250) — matrix operand order repaired and checked by reading, not compiled, not run on hardware
//
#include <hip/hip_runtime.h>
#include <hip/hip_bf16.h>
#include <math.h>

// Problem constants (from reference)
#define BB    8
#define NN    16384
#define CIN   256
#define TD    512
#define DD    64
#define MQ    16
#define KTOP  64
#define LLEN  80
#define NH    4
#define DH    16

typedef __attribute__((ext_vector_type(16))) __bf16        v16bf;
typedef __attribute__((ext_vector_type(8)))  float         v8f;
typedef __attribute__((ext_vector_type(4)))  unsigned int  u32x4;
typedef __attribute__((ext_vector_type(8)))  unsigned int  u32x8;

struct U32x8 { u32x4 lo, hi; };

static __device__ __forceinline__ unsigned short f2bf(float f) {
  unsigned int x = __builtin_bit_cast(unsigned int, f);
  x += 0x7FFFu + ((x >> 16) & 1u);           // round-to-nearest-even
  return (unsigned short)(x >> 16);
}
static __device__ __forceinline__ unsigned pk2(float a, float b) {
  return (unsigned)f2bf(a) | ((unsigned)f2bf(b) << 16);
}
static __device__ __forceinline__ v16bf mkfrag(u32x4 lo, u32x4 hi) {
  U32x8 t; t.lo = lo; t.hi = hi;
  return __builtin_bit_cast(v16bf, t);
}
static __device__ __forceinline__ float gelu_exact(float x) {
  return 0.5f * x * (1.0f + erff(x * 0.70710678118654752f));
}
// Scheduling fence: memory ops above cannot sink below (batches DS waits).
static __device__ __forceinline__ void sched_fence() {
  asm volatile("" ::: "memory");
}

// ---- CDNA5 async global->LDS (per-lane, ASYNCcnt) ------------------------
static __device__ __forceinline__ void async_b128(unsigned lds_off, const void* gptr) {
  asm volatile("global_load_async_to_lds_b128 %0, %1, off" :: "v"(lds_off), "v"(gptr) : "memory");
}
static __device__ __forceinline__ void wait_async0() {
  asm volatile("s_wait_asynccnt 0x0" ::: "memory");
}

// ---- CDNA5 Tensor Data Mover: 2D bf16/2-byte tile, global -> LDS ---------
// D# group0/group1 per CDNA5 ISA.
// pad_interval_log: pad every 2^(p+1) DWORDs; pad_amount_m1: (n+1) DWORDs pad.
static __device__ __forceinline__ void tdm_load_2d_b16(
    unsigned lds_off, const void* gptr,
    unsigned tensor_d0, unsigned tensor_d1, unsigned d0_stride,
    unsigned tile_d0, unsigned tile_d1,
    unsigned pad_interval_log, unsigned pad_amount_m1) {
  unsigned long long ga = (unsigned long long)(size_t)gptr;
  u32x4 g0;
  g0[0] = 1u;                                      // count=1 (valid user descriptor)
  g0[1] = lds_off;                                 // lds_addr (bytes)
  g0[2] = (unsigned)ga;                            // global_addr[31:0]
  g0[3] = (unsigned)(ga >> 32) | (2u << 30);       // global_addr[56:32] | type=2
  u32x8 g1;
  g1[0] = (1u << 16)                               // data_size = 2 bytes
        | (1u << 20)                               // pad_enable
        | (pad_interval_log << 22)
        | (pad_amount_m1 << 25);
  g1[1] = (tensor_d0 & 0xFFFFu) << 16;             // tensor_dim0[15:0] @ bits 63:48
  g1[2] = ((tensor_d0 >> 16) & 0xFFFFu)            // tensor_dim0[31:16]
        | ((tensor_d1 & 0xFFFFu) << 16);           // tensor_dim1[15:0]
  g1[3] = ((tensor_d1 >> 16) & 0xFFFFu)            // tensor_dim1[31:16]
        | (tile_d0 << 16);                         // tile_dim0
  g1[4] = tile_d1 & 0xFFFFu;                       // tile_dim1 (tile_dim2 = 0)
  g1[5] = d0_stride;                               // tensor_dim0_stride[31:0]
  g1[6] = 0;                                       // stride[47:32] | dim1_stride lo (unused, 2D)
  g1[7] = 0;
  asm volatile("tensor_load_to_lds %0, %1" :: "s"(g0), "s"(g1) : "memory");
}

// ---------------------------------------------------------------- K0: fp32 -> bf16 weight copies
__global__ void k0_convert(const float* __restrict__ W1, const float* __restrict__ W2,
                           unsigned short* __restrict__ w1bf, unsigned short* __restrict__ w2bf) {
  int i = blockIdx.x * 256 + threadIdx.x;
  if (i < TD * CIN) w1bf[i] = f2bf(W1[i]);
  if (i < DD * TD)  w2bf[i] = f2bf(W2[i]);
}

// ---------------------------------------------------------------- K_txt: txt = l2norm(text[:,0] @ W2^T + b2)
__global__ void k_txt(const float* __restrict__ text, const float* __restrict__ W2,
                      const float* __restrict__ b2, float* __restrict__ txtn) {
  __shared__ float red[64];
  int b = blockIdx.x, d = threadIdx.x;
  const float* t = text + (size_t)b * 4 * TD;     // pooled = row 0
  float acc = b2[d];
  for (int k = 0; k < TD; ++k) acc += t[k] * W2[(size_t)d * TD + k];
  float v = acc + 1e-8f;
  red[d] = v * v; __syncthreads();
  for (int s = 32; s > 0; s >>= 1) { if (d < s) red[d] += red[d + s]; __syncthreads(); }
  float nrm = sqrtf(red[0]);
  txtn[b * DD + d] = v / fmaxf(nrm, 1e-12f);
}

// ---------------------------------------------------------------- K1: fused  act = gelu(X@W1^T+b1)@W2^T+b2
// X tile staged by async global->LDS (ASYNCcnt); W1/W2 slices DMA'd by the
// Tensor Data Mover (TENSORcnt, padded LDS rows); GEMMs on v_wmma bf16.
__global__ void __launch_bounds__(128) k1_mlp(
    const float* __restrict__ X, const unsigned short* __restrict__ w1bf,
    const unsigned short* __restrict__ w2bf, const float* __restrict__ b1,
    const float* __restrict__ b2, float* __restrict__ act) {
  __shared__ float          lxf[64][260];   // X tile fp32 (async-loaded), padded rows
  __shared__ unsigned short lw1[32][264];   // W1 slice: 32 proj cols x 256 K (TDM pad 16B/row)
  __shared__ unsigned short lw2[64][40];    // W2 slice: 64 out cols x 32 K  (TDM pad 16B/row)
  __shared__ unsigned short lpr[4][16][40]; // per-wave gelu(proj) staging, 16x32

  const int tid  = threadIdx.x;
  const int w    = tid >> 5;
  const int lane = tid & 31;
  const int r0   = blockIdx.x * 64;

  // Stage X tile (64 rows x 256 fp32) via async global->LDS, no VGPR staging
  for (int g = tid; g < 64 * 64; g += 128) {
    int r = g >> 6, c4 = (g & 63) << 2;
    async_b128((unsigned)(size_t)&lxf[r][c4], X + (size_t)(r0 + r) * CIN + c4);
  }
  wait_async0();
  __syncthreads();

  // Build bf16 A fragments in registers straight from the fp32 LDS tile.
  // Dense 16-bit A 16x32 layout: lane l -> M=l&15 ; K = (l>=16?8:0) + (h<8 ? h : 16+h-8)
  const int m     = (w << 4) + (lane & 15);
  const int abase = (lane >> 4) << 3;        // 0 or 8
  v16bf afrag[8];
#pragma unroll
  for (int kc = 0; kc < 8; ++kc) {
    float4 fa = *(const float4*)&lxf[m][kc * 32 + abase];
    float4 fb = *(const float4*)&lxf[m][kc * 32 + abase + 4];
    float4 fc = *(const float4*)&lxf[m][kc * 32 + abase + 16];
    float4 fd = *(const float4*)&lxf[m][kc * 32 + abase + 20];
    u32x4 lo, hi;
    lo[0] = pk2(fa.x, fa.y); lo[1] = pk2(fa.z, fa.w);
    lo[2] = pk2(fb.x, fb.y); lo[3] = pk2(fb.z, fb.w);
    hi[0] = pk2(fc.x, fc.y); hi[1] = pk2(fc.z, fc.w);
    hi[2] = pk2(fd.x, fd.y); hi[3] = pk2(fd.z, fd.w);
    afrag[kc] = mkfrag(lo, hi);
  }

  // B layout: lane l -> N=l&15 ; K = (l>=16?8:0) + (h<8 ? h : 16+h-8)
  const int bbase = (lane >> 4) << 3;        // 0 or 8
  const unsigned lw1_off = (unsigned)(size_t)&lw1[0][0];
  const unsigned lw2_off = (unsigned)(size_t)&lw2[0][0];

  v8f acc[4];
#pragma unroll
  for (int t = 0; t < 4; ++t) {
    float bv = b2[t * 16 + (lane & 15)];
#pragma unroll
    for (int i = 0; i < 8; ++i) acc[t][i] = bv;
  }

  for (int j2 = 0; j2 < 16; ++j2) {          // 32 proj cols per step
    // Bias scalars for this slice: issued early, consumed only in the epilogue
    float bv0 = b1[j2 * 32 + (lane & 15)];
    float bv1 = b1[j2 * 32 + 16 + (lane & 15)];
    __syncthreads();                          // previous slice fully consumed
    if (w == 0) {
      // W1 slice: 32 rows x 256 bf16, row=512B=128 DW -> pad_interval_log=6, pad 4 DW
      tdm_load_2d_b16(lw1_off, w1bf + (size_t)j2 * 32 * CIN,
                      CIN, TD, CIN, CIN, 32, 6, 3);
      // W2 slice: 64 rows x 32 bf16, row=64B=16 DW -> pad_interval_log=3, pad 4 DW
      tdm_load_2d_b16(lw2_off, w2bf + (size_t)j2 * 32,
                      TD, DD, TD, 32, DD, 3, 3);
      __builtin_amdgcn_s_wait_tensorcnt(0);
    }
    __syncthreads();

    // GEMM1: two 16x16 proj tiles (K=256); C seeded with inline 0, bias folded
    // into the GELU epilogue so no load feeds the WMMA chain.
#pragma unroll
    for (int t = 0; t < 2; ++t) {
      int nl = t * 16 + (lane & 15);
      u32x4 blo[8], bhi[8];
#pragma unroll
      for (int kc = 0; kc < 8; ++kc) {        // batch all 16 ds_load_b128
        blo[kc] = *(const u32x4*)&lw1[nl][kc * 32 + bbase];
        bhi[kc] = *(const u32x4*)&lw1[nl][kc * 32 + bbase + 16];
      }
      sched_fence();                          // loads issue above; single DS drain
      v8f pa{};
#pragma unroll
      for (int kc = 0; kc < 8; ++kc) {
        pa = __builtin_amdgcn_wmma_f32_16x16x32_bf16(false, afrag[kc], false,
                                                     mkfrag(blo[kc], bhi[kc]),
                                                     (short)0, pa, false, false);
      }
      float bv = t ? bv1 : bv0;
      int prow = (lane >> 4) << 3;   // C layout: M = v + 8*(lane>=16)
#pragma unroll
      for (int v = 0; v < 8; ++v)
        lpr[w][prow + v][t * 16 + (lane & 15)] = f2bf(gelu_exact(pa[v] + bv));
    }

    // GEMM2: act(16x64) += gelu(proj)(16x32) @ W2slice(32x64)
    u32x4 alo = *(const u32x4*)&lpr[w][lane & 15][abase];
    u32x4 ahi = *(const u32x4*)&lpr[w][lane & 15][abase + 16];
    u32x4 clo[4], chi[4];
#pragma unroll
    for (int t = 0; t < 4; ++t) {
      int col = t * 16 + (lane & 15);
      clo[t] = *(const u32x4*)&lw2[col][bbase];
      chi[t] = *(const u32x4*)&lw2[col][bbase + 16];
    }
    sched_fence();                            // batch A2 + 8 B-frag loads
    v16bf a2 = mkfrag(alo, ahi);
#pragma unroll
    for (int t = 0; t < 4; ++t) {
      acc[t] = __builtin_amdgcn_wmma_f32_16x16x32_bf16(false, a2, false,
                                                       mkfrag(clo[t], chi[t]),
                                                       (short)0, acc[t], false, false);
    }
  }

  int rw = r0 + (w << 4) + ((lane >> 4) << 3);
#pragma unroll
  for (int t = 0; t < 4; ++t)
#pragma unroll
    for (int v = 0; v < 8; ++v)
      act[(size_t)(rw + v) * DD + t * 16 + (lane & 15)] = acc[t][v];
}

// ---------------------------------------------------------------- K2: cosine scores (wave per row)
__global__ void k2_scores(const float* __restrict__ act, const float* __restrict__ txtn,
                          float* __restrict__ scores) {
  int row  = blockIdx.x * 8 + (threadIdx.x >> 5);
  int lane = threadIdx.x & 31;
  int b    = row >> 14;
  float a0 = act[(size_t)row * DD + 2 * lane]     + 1e-8f;
  float a1 = act[(size_t)row * DD + 2 * lane + 1] + 1e-8f;
  float t0 = txtn[b * DD + 2 * lane];
  float t1 = txtn[b * DD + 2 * lane + 1];
  float ss = a0 * a0 + a1 * a1;
  float dd = a0 * t0 + a1 * t1;
  for (int o = 16; o; o >>= 1) { ss += __shfl_xor(ss, o, 32); dd += __shfl_xor(dd, o, 32); }
  if (lane == 0) scores[row] = dd / fmaxf(sqrtf(ss), 1e-12f);
}

// ---------------------------------------------------------------- K3: deterministic top-64 + ascending sort
__global__ void k3_topk(const float* __restrict__ scores, int* __restrict__ idx) {
  __shared__ float rv[256];
  __shared__ int   ri[256];
  __shared__ int   chosen[KTOP];
  int b = blockIdx.x, tid = threadIdx.x;
  float loc[64];
  for (int s = 0; s < 64; ++s) loc[s] = scores[(size_t)b * NN + s * 256 + tid];
  for (int it = 0; it < KTOP; ++it) {
    float best = -INFINITY; int bi = NN;
    for (int s = 0; s < 64; ++s) {           // increasing i -> '>' keeps lowest index on ties
      float v = loc[s];
      int i = s * 256 + tid;
      if (v > best) { best = v; bi = i; }
    }
    rv[tid] = best; ri[tid] = bi;
    __syncthreads();
    for (int s = 128; s > 0; s >>= 1) {
      if (tid < s) {
        if (rv[tid + s] > rv[tid] || (rv[tid + s] == rv[tid] && ri[tid + s] < ri[tid])) {
          rv[tid] = rv[tid + s]; ri[tid] = ri[tid + s];
        }
      }
      __syncthreads();
    }
    int wi = ri[0];
    if (tid == 0) chosen[it] = wi;
    if ((wi & 255) == tid) loc[wi >> 8] = -INFINITY;
    __syncthreads();
  }
  if (tid == 0) {
    for (int i = 1; i < KTOP; ++i) {
      int key = chosen[i], j = i - 1;
      while (j >= 0 && chosen[j] > key) { chosen[j + 1] = chosen[j]; --j; }
      chosen[j + 1] = key;
    }
    for (int i = 0; i < KTOP; ++i) idx[b * KTOP + i] = chosen[i];
  }
}

// ---------------------------------------------------------------- K4: LN + 4-head attention over 80 tokens (fp32)
__global__ void __launch_bounds__(256) k4_attn(
    const float* __restrict__ act, const int* __restrict__ idx,
    const float* __restrict__ mq, const float* __restrict__ lng,
    const float* __restrict__ lnb, const float* __restrict__ Wqkv,
    const float* __restrict__ bqkv, const float* __restrict__ Wo,
    const float* __restrict__ bo, float* __restrict__ ao_ws,
    float* __restrict__ enh_ws) {
  __shared__ float x[LLEN][DD];
  __shared__ float kk[LLEN][DD];
  __shared__ float vv[LLEN][DD];
  int b = blockIdx.x, tid = threadIdx.x;
  int w = tid >> 5, lane = tid & 31;

  // LayerNorm of comb rows (comb recomputed from sources; not stored)
  for (int l = w; l < LLEN; l += 8) {
    int c0 = 2 * lane, c1 = 2 * lane + 1;
    float v0, v1;
    if (l < MQ) { v0 = mq[l * DD + c0]; v1 = mq[l * DD + c1]; }
    else {
      int r = idx[b * KTOP + (l - MQ)];
      const float* ar = act + ((size_t)b * NN + r) * DD;
      v0 = ar[c0]; v1 = ar[c1];
    }
    float s = v0 + v1;
    for (int o = 16; o; o >>= 1) s += __shfl_xor(s, o, 32);
    float mu = s * (1.0f / 64.0f);
    float d0 = v0 - mu, d1 = v1 - mu;
    float vs = d0 * d0 + d1 * d1;
    for (int o = 16; o; o >>= 1) vs += __shfl_xor(vs, o, 32);
    float rs = rsqrtf(vs * (1.0f / 64.0f) + 1e-5f);
    x[l][c0] = d0 * rs * lng[c0] + lnb[c0];
    x[l][c1] = d1 * rs * lng[c1] + lnb[c1];
  }
  __syncthreads();

  // K and V projections (Wqkv rows 64..191)
  for (int o = tid; o < LLEN * 128; o += 256) {
    int l = o >> 7, j = o & 127;
    int wr = 64 + j;
    float s = bqkv[wr];
    for (int d = 0; d < DD; ++d) s += x[l][d] * Wqkv[(size_t)wr * DD + d];
    if (j < 64) kk[l][j] = s; else vv[l][j - 64] = s;
  }
  __syncthreads();

  // One wave per (head, query-row); q recomputed on the fly
  for (int task = w; task < NH * LLEN; task += 8) {
    int h = task / LLEN, qi = task % LLEN;
    int hd0 = h * DH;
    float qreg = 0.0f;
    if (lane < DH) {
      int wr = hd0 + lane;
      float s = bqkv[wr];
      for (int d = 0; d < DD; ++d) s += x[qi][d] * Wqkv[(size_t)wr * DD + d];
      qreg = s;
    }
    int k0 = lane, k1 = lane + 32, k2 = lane + 64;
    bool v2 = (k2 < LLEN);
    int k2i = v2 ? k2 : 0;
    float s0 = 0.0f, s1 = 0.0f, s2 = 0.0f;
    for (int d = 0; d < DH; ++d) {
      float qv = __shfl(qreg, d, 32);
      s0 += qv * kk[k0][hd0 + d];
      s1 += qv * kk[k1][hd0 + d];
      s2 += qv * kk[k2i][hd0 + d];
    }
    s0 *= 0.25f; s1 *= 0.25f; s2 = v2 ? s2 * 0.25f : -INFINITY;
    float mx = fmaxf(fmaxf(s0, s1), s2);
    for (int o = 16; o; o >>= 1) mx = fmaxf(mx, __shfl_xor(mx, o, 32));
    float p0 = expf(s0 - mx), p1 = expf(s1 - mx);
    float p2 = v2 ? expf(s2 - mx) : 0.0f;
    float ps = p0 + p1 + p2;
    for (int o = 16; o; o >>= 1) ps += __shfl_xor(ps, o, 32);
    float inv = 1.0f / ps;
    for (int d = 0; d < DH; ++d) {
      float od = p0 * vv[k0][hd0 + d] + p1 * vv[k1][hd0 + d] + p2 * vv[k2i][hd0 + d];
      for (int o = 16; o; o >>= 1) od += __shfl_xor(od, o, 32);
      if (lane == 0) ao_ws[((size_t)b * LLEN + qi) * DD + hd0 + d] = od * inv;
    }
  }
  __threadfence_block();
  __syncthreads();

  // Output projection + residual with comb (recomputed)
  for (int o = tid; o < LLEN * DD; o += 256) {
    int l = o >> 6, c = o & 63;
    float combv;
    if (l < MQ) combv = mq[l * DD + c];
    else { int r = idx[b * KTOP + (l - MQ)]; combv = act[((size_t)b * NN + r) * DD + c]; }
    float s = bo[c];
    const float* aor = ao_ws + ((size_t)b * LLEN + l) * DD;
    for (int d = 0; d < DD; ++d) s += aor[d] * Wo[(size_t)c * DD + d];
    enh_ws[((size_t)b * LLEN + l) * DD + c] = combv + s;
  }
}

// ---------------------------------------------------------------- K5: out = image + gamma*bup (broadcast)
__global__ void k5_base(const float* __restrict__ img, const float* __restrict__ bup,
                        const float* __restrict__ gammap, float* __restrict__ out) {
  size_t i = (size_t)blockIdx.x * 256 + threadIdx.x;   // float4 index
  float g = gammap[0];
  float4 v = ((const float4*)img)[i];
  float4 bb = ((const float4*)bup)[(int)(i & 63)];
  float4 o;
  o.x = v.x + g * bb.x; o.y = v.y + g * bb.y;
  o.z = v.z + g * bb.z; o.w = v.w + g * bb.w;
  ((float4*)out)[i] = o;
}

// ---------------------------------------------------------------- K6: out[b, idx[b,k], :] += gamma * (enh_sparse @ Wup^T)
__global__ void k6_scatter(const float* __restrict__ enh_ws, const int* __restrict__ idx,
                           const float* __restrict__ Wup, const float* __restrict__ gammap,
                           float* __restrict__ out) {
  __shared__ float e[DD];
  int blk = blockIdx.x, tid = threadIdx.x;
  int b = blk >> 6, kq = blk & 63;
  if (tid < 16) {      // async global->LDS fetch of the enhanced row (64 floats)
    async_b128((unsigned)(size_t)&e[tid * 4],
               enh_ws + ((size_t)b * LLEN + MQ + kq) * DD + tid * 4);
  }
  wait_async0();
  __syncthreads();
  int r = idx[b * KTOP + kq];
  float g = gammap[0];
  float s = 0.0f;
  for (int d = 0; d < DD; ++d) s += e[d] * Wup[(size_t)tid * DD + d];
  out[((size_t)b * NN + r) * CIN + tid] += g * s;
}

// ---------------------------------------------------------------- launch
extern "C" void kernel_launch(void* const* d_in, const int* in_sizes, int n_in,
                              void* d_out, int out_size, void* d_ws, size_t ws_size,
                              hipStream_t stream) {
  const float* img   = (const float*)d_in[0];
  const float* text  = (const float*)d_in[1];
  const float* W1    = (const float*)d_in[2];
  const float* b1    = (const float*)d_in[3];
  const float* W2    = (const float*)d_in[4];
  const float* b2    = (const float*)d_in[5];
  const float* mq    = (const float*)d_in[6];
  const float* lng   = (const float*)d_in[7];
  const float* lnb   = (const float*)d_in[8];
  const float* Wqkv  = (const float*)d_in[9];
  const float* bqkv  = (const float*)d_in[10];
  const float* Wo    = (const float*)d_in[11];
  const float* bo    = (const float*)d_in[12];
  const float* Wup   = (const float*)d_in[13];
  const float* bup   = (const float*)d_in[14];
  const float* gamma = (const float*)d_in[15];
  float* out = (float*)d_out;

  // workspace layout (bytes); total ~34.7 MB
  char* ws = (char*)d_ws;
  unsigned short* w1bf = (unsigned short*)(ws + 0);            // 512*256*2   = 262144
  unsigned short* w2bf = (unsigned short*)(ws + 262144);       // 64*512*2    = 65536
  float* txtn   = (float*)(ws + 327680);                       // 8*64*4      = 2048
  int*   idx    = (int*)  (ws + 329728);                       // 8*64*4      = 2048
  float* ao_ws  = (float*)(ws + 331776);                       // 8*80*64*4   = 163840
  float* enh_ws = (float*)(ws + 495616);                       // 8*80*64*4   = 163840
  float* scores = (float*)(ws + 659456);                       // 8*16384*4   = 524288
  float* act    = (float*)(ws + 1183744);                      // 8*16384*64*4 = 33554432

  k0_convert<<<512, 256, 0, stream>>>(W1, W2, w1bf, w2bf);
  k_txt<<<BB, 64, 0, stream>>>(text, W2, b2, txtn);
  k1_mlp<<<(BB * NN) / 64, 128, 0, stream>>>(img, w1bf, w2bf, b1, b2, act);
  k2_scores<<<(BB * NN) / 8, 256, 0, stream>>>(act, txtn, scores);
  k3_topk<<<BB, 256, 0, stream>>>(scores, idx);
  k4_attn<<<BB, 256, 0, stream>>>(act, idx, mq, lng, lnb, Wqkv, bqkv, Wo, bo, ao_ws, enh_ws);
  k5_base<<<(BB * NN * CIN / 4) / 256, 256, 0, stream>>>(img, bup, gamma, out);
  k6_scatter<<<BB * KTOP, 256, 0, stream>>>(enh_ws, idx, Wup, gamma, out);
}
